// AE_30374008717946
// MI455X (gfx1250) — matrix operand order repaired and checked by reading, not compiled, not run on hardware
//
#include <hip/hip_runtime.h>
#include <hip/hip_bf16.h>

// ---------------------------------------------------------------------------
// LSTM autoencoder for MI455X (gfx1250, wave32, WMMA bf16 -> f32 accumulate)
//
// Strategy:
//  * Convert all weights once to bf16, pre-transposed to [N][K] so B-matrix
//    WMMA fragments are contiguous 32B loads per lane. Weights (~15MB bf16)
//    stay resident in the 192MB L2 across the whole scan.
//  * Persistent scan kernel: grid (8 n-groups x 4 batch-groups), 4 waves/WG.
//    Each wave owns a 16-column group of h across all 4 gates, so gate math
//    and the c/h update are pure per-lane register ops. c lives in VGPRs for
//    the entire T=1024 loop.
//  * Input projection is fused into the scan: K = Din + 512, weights
//    [Wk;Wr] concatenated, bias folded into the accumulator init.
//  * h is exchanged between the 8 n-group WGs of a batch group through a
//    double-buffered global (L2-hot) buffer with one atomic phase barrier
//    per step (s_sleep backoff).
//  * Dense layers: parallel WMMA GEMM over the 65536x512 activations.
// ---------------------------------------------------------------------------

typedef __bf16 bf16_t;
typedef __attribute__((ext_vector_type(16))) __bf16 v16bf;
typedef __attribute__((ext_vector_type(8)))  __bf16 v8bf;
typedef __attribute__((ext_vector_type(8)))  float  v8f;

#define B_    64
#define T_    1024
#define D_    32
#define H_    512
#define FOURH 2048

// ---- WMMA helpers ---------------------------------------------------------

__device__ __forceinline__ v8f wmma_bf16(v16bf a, v16bf b, v8f c) {
  // D(f32 16x16) = A(16x32 bf16) * B(32x16 bf16) + C
  return __builtin_amdgcn_wmma_f32_16x16x32_bf16(
      /*neg_a=*/false, a, /*neg_b=*/false, b,
      /*c_mod=*/(short)0, c, /*reuse_a=*/false, /*reuse_b=*/false);
}

// A fragment, this file's contiguous k order (as in B): lanes 0-15 = rows M=0..15 holding
// K = {0..15}; lanes 16-31 same rows, K = {16..31}.
// rowp points at element (row = lane&15, k0) of a row-major bf16 matrix.
__device__ __forceinline__ v16bf load_a_frag(const bf16_t* __restrict__ rowp,
                                             int half) {
  v8bf lo = *(const v8bf*)(rowp + half * 16);
  v8bf hi = *(const v8bf*)(rowp + half * 16 + 8);
  v16bf a;
#pragma unroll
  for (int i = 0; i < 8; ++i) { a[i] = lo[i]; a[i + 8] = hi[i]; }
  return a;
}

// B fragment, this file's contiguous k order (as in A): lane (l&15) = column N, (l>>4) selects
// K 0..15 vs 16..31. With weights stored transposed as WT[n][k] (row-major
// in k), each lane reads 16 contiguous bf16 (32 bytes).
__device__ __forceinline__ v16bf load_b_frag(const bf16_t* __restrict__ p) {
  return *(const v16bf*)p;
}

__device__ __forceinline__ float sigmoidf_(float x) {
  return 1.0f / (1.0f + __expf(-x));
}

// ---- prep kernels ---------------------------------------------------------

__global__ void init_counters_kernel(unsigned* cnt, int n) {
  int i = blockIdx.x * blockDim.x + threadIdx.x;
  if (i < n) cnt[i] = 0u;
}

__global__ void convert_bf16_kernel(const float* __restrict__ src,
                                    bf16_t* __restrict__ dst, int n) {
  int i = blockIdx.x * blockDim.x + threadIdx.x;
  if (i < n) dst[i] = (bf16_t)src[i];
}

// WT[n][k], n in [0,4H), k in [0,Din+H): k < Din -> Wk[k][n], else Wr[k-Din][n]
__global__ void prep_lstm_w_kernel(const float* __restrict__ Wk,
                                   const float* __restrict__ Wr, int Din,
                                   bf16_t* __restrict__ WT) {
  int K = Din + H_;
  int total = FOURH * K;
  for (int idx = blockIdx.x * blockDim.x + threadIdx.x; idx < total;
       idx += gridDim.x * blockDim.x) {
    int n = idx / K;
    int k = idx - n * K;
    float v = (k < Din) ? Wk[(size_t)k * FOURH + n]
                        : Wr[(size_t)(k - Din) * FOURH + n];
    WT[idx] = (bf16_t)v;
  }
}

// dense W is [K=512][N]; WT[n][k] = W[k][n]
__global__ void prep_dense_w_kernel(const float* __restrict__ W, int N,
                                    bf16_t* __restrict__ WT) {
  int total = N * H_;
  for (int idx = blockIdx.x * blockDim.x + threadIdx.x; idx < total;
       idx += gridDim.x * blockDim.x) {
    int n = idx / H_;
    int k = idx - n * H_;
    WT[idx] = (bf16_t)W[(size_t)k * N + n];
  }
}

// ---- cross-WG phase barrier (per batch group, 8 member WGs) ---------------

__device__ __forceinline__ void arrive_wait(unsigned* cnt, unsigned target) {
  __syncthreads();                 // all waves in WG finished their stores
  if (threadIdx.x == 0) {
    __threadfence();               // release: make h stores device-visible
    atomicAdd(cnt, 1u);
    while (__hip_atomic_load(cnt, __ATOMIC_RELAXED,
                             __HIP_MEMORY_SCOPE_AGENT) < target) {
      __builtin_amdgcn_s_sleep(2);
    }
  }
  __syncthreads();
  __threadfence();                 // acquire: see peers' h stores
}

// ---- persistent LSTM scan kernel -----------------------------------------
// grid = (8 n-groups, 4 batch-groups), block = 128 (4 waves).
// Wave w owns h columns [ng*64 + w*16, +16) for all four gates.
// in : [B][T][Din] bf16      WT : [4H][Din+512] bf16   bias: [4H] f32
// out: [B][T][512] bf16      hbuf: double buffer 2 x [64][512] bf16
__global__ void __launch_bounds__(128) lstm_layer_kernel(
    const bf16_t* __restrict__ in, int Din,
    const bf16_t* __restrict__ WT, const float* __restrict__ bias,
    bf16_t* __restrict__ out, bf16_t* __restrict__ hbuf,
    unsigned* __restrict__ cnt) {
  const int lane = threadIdx.x & 31;
  const int wave = threadIdx.x >> 5;   // 0..3
  const int ng   = blockIdx.x;         // 0..7  (h-column group of 64)
  const int bg   = blockIdx.y;         // 0..3  (batch group of 16 rows)
  const int b0   = bg * 16;
  const int cg0  = ng * 64 + wave * 16;     // h column base for this wave
  const int K    = Din + H_;
  const int r15  = lane & 15;
  const int half = lane >> 4;
  unsigned* myCnt = cnt + bg;

  // zero our chunk of h double-buffer 0 (initial hidden state)
  for (int i = threadIdx.x; i < 16 * 64; i += blockDim.x) {
    int rr = i >> 6, cc = i & 63;
    hbuf[(size_t)(b0 + rr) * H_ + ng * 64 + cc] = (bf16_t)0.0f;
  }
  arrive_wait(myCnt, 8u * 1u);  // phase 0: h(t=-1) == 0 visible everywhere

  // bias per gate for this lane's column (uniform over rows)
  float bv[4];
#pragma unroll
  for (int g = 0; g < 4; ++g) bv[g] = bias[g * H_ + cg0 + r15];

  v8f c = {};  // cell state fragment lives in VGPRs for the whole scan

  for (int t = 0; t < T_; ++t) {
    const bf16_t* hread  = hbuf + (size_t)(t & 1) * (64 * H_);
    bf16_t*       hwrite = hbuf + (size_t)((t + 1) & 1) * (64 * H_);

    v8f acc[4];
#pragma unroll
    for (int g = 0; g < 4; ++g) {
#pragma unroll
      for (int r = 0; r < 8; ++r) acc[g][r] = bv[g];
    }

    // ---- z += x_t @ Wk : K-slabs over the input columns ----
    const bf16_t* arow = in + ((size_t)(b0 + r15) * T_ + t) * Din;
    for (int k0 = 0; k0 < Din; k0 += 32) {
      v16bf a = load_a_frag(arow + k0, half);
#pragma unroll
      for (int g = 0; g < 4; ++g) {
        const bf16_t* bp =
            WT + (size_t)(g * H_ + cg0 + r15) * K + k0 + half * 16;
        acc[g] = wmma_bf16(a, load_b_frag(bp), acc[g]);
      }
    }

    // ---- z += h_{t-1} @ Wr : K-slabs over the 512 hidden columns ----
    const bf16_t* hrow = hread + (size_t)(b0 + r15) * H_;
    for (int k0 = 0; k0 < H_; k0 += 32) {
      v16bf a = load_a_frag(hrow + k0, half);
#pragma unroll
      for (int g = 0; g < 4; ++g) {
        const bf16_t* bp =
            WT + (size_t)(g * H_ + cg0 + r15) * K + (Din + k0) + half * 16;
        acc[g] = wmma_bf16(a, load_b_frag(bp), acc[g]);
      }
    }

    // ---- gates + state update, pure per-lane register math ----
#pragma unroll
    for (int r = 0; r < 8; ++r) {
      float iv = sigmoidf_(acc[0][r]);
      float fv = sigmoidf_(acc[1][r]);
      float gv = tanhf(acc[2][r]);
      float ov = sigmoidf_(acc[3][r]);
      float cv = fv * c[r] + iv * gv;
      c[r] = cv;
      float hv = ov * tanhf(cv);
      int row = b0 + half * 8 + r;   // C/D layout: lanes>=16 hold rows 8..15
      int col = cg0 + r15;
      bf16_t hb = (bf16_t)hv;
      hwrite[(size_t)row * H_ + col] = hb;
      out[((size_t)row * T_ + t) * H_ + col] = hb;
    }

    arrive_wait(myCnt, 8u * (unsigned)(t + 2));
  }
}

// ---- dense (time-distributed) WMMA GEMM ----------------------------------
// in: [BT][512] bf16, WT: [Nout][512] bf16, bias: [Nout] f32
// outf: [BT][Nout] f32, outbf (optional): [BT][Nout] bf16
__global__ void __launch_bounds__(256) dense_kernel(
    const bf16_t* __restrict__ in, const bf16_t* __restrict__ WT,
    const float* __restrict__ bias, int Nout,
    float* __restrict__ outf, bf16_t* __restrict__ outbf) {
  const int lane = threadIdx.x & 31;
  const int wave = threadIdx.x >> 5;  // 0..7
  const int r15  = lane & 15;
  const int half = lane >> 4;
  const int n0 = blockIdx.x * 16;
  const int m0 = (blockIdx.y * 8 + wave) * 16;

  float bv = bias[n0 + r15];
  v8f acc;
#pragma unroll
  for (int r = 0; r < 8; ++r) acc[r] = bv;

  const bf16_t* arow = in + (size_t)(m0 + r15) * H_;
  const bf16_t* brow = WT + (size_t)(n0 + r15) * H_;
#pragma unroll
  for (int k0 = 0; k0 < H_; k0 += 32) {
    v16bf a = load_a_frag(arow + k0, half);
    v16bf b = load_b_frag(brow + k0 + half * 16);
    acc = wmma_bf16(a, b, acc);
  }

  int col = n0 + r15;
#pragma unroll
  for (int r = 0; r < 8; ++r) {
    int row = m0 + half * 8 + r;
    float v = acc[r];
    outf[(size_t)row * Nout + col] = v;
    if (outbf) outbf[(size_t)row * Nout + col] = (bf16_t)v;
  }
}

// ---- host-side orchestration ---------------------------------------------

extern "C" void kernel_launch(void* const* d_in, const int* in_sizes, int n_in,
                              void* d_out, int out_size, void* d_ws,
                              size_t ws_size, hipStream_t stream) {
  const float* x      = (const float*)d_in[0];
  const float* enc_k0 = (const float*)d_in[1];
  const float* enc_r0 = (const float*)d_in[2];
  const float* enc_b0 = (const float*)d_in[3];
  const float* enc_k1 = (const float*)d_in[4];
  const float* enc_r1 = (const float*)d_in[5];
  const float* enc_b1 = (const float*)d_in[6];
  const float* enc_dw = (const float*)d_in[7];
  const float* enc_db = (const float*)d_in[8];
  const float* dec_k0 = (const float*)d_in[9];
  const float* dec_r0 = (const float*)d_in[10];
  const float* dec_r1 = (const float*)d_in[13];
  const float* dec_k1 = (const float*)d_in[12];
  const float* dec_b0 = (const float*)d_in[11];
  const float* dec_b1 = (const float*)d_in[14];
  const float* dec_dw = (const float*)d_in[15];
  const float* dec_db = (const float*)d_in[16];

  float* outf = (float*)d_out;
  const size_t enc_elems = (size_t)B_ * T_ * H_;  // 33,554,432

  // workspace carve-up (256B aligned)
  char* base = (char*)d_ws;
  size_t off = 0;
  auto carve = [&](size_t bytes) -> void* {
    void* p = base + off;
    off = (off + bytes + 255) & ~(size_t)255;
    return p;
  };
  unsigned* cnt  = (unsigned*)carve(16 * sizeof(unsigned));
  bf16_t* xbf    = (bf16_t*)carve((size_t)B_ * T_ * D_ * 2);
  bf16_t* seqA   = (bf16_t*)carve((size_t)B_ * T_ * H_ * 2);
  bf16_t* seqB   = (bf16_t*)carve((size_t)B_ * T_ * H_ * 2);
  bf16_t* hbuf   = (bf16_t*)carve((size_t)2 * 64 * H_ * 2);
  bf16_t* WT0    = (bf16_t*)carve((size_t)FOURH * (D_ + H_) * 2);
  bf16_t* WT1    = (bf16_t*)carve((size_t)FOURH * (H_ + H_) * 2);
  bf16_t* WT2    = (bf16_t*)carve((size_t)FOURH * (H_ + H_) * 2);
  bf16_t* WT3    = (bf16_t*)carve((size_t)FOURH * (H_ + H_) * 2);
  bf16_t* WTde   = (bf16_t*)carve((size_t)H_ * H_ * 2);
  bf16_t* WTdd   = (bf16_t*)carve((size_t)D_ * H_ * 2);
  (void)ws_size; (void)n_in; (void)in_sizes; (void)out_size;

  // prep: counters, bf16 input, bf16 transposed weights
  init_counters_kernel<<<1, 32, 0, stream>>>(cnt, 16);
  {
    int n = B_ * T_ * D_;
    convert_bf16_kernel<<<(n + 255) / 256, 256, 0, stream>>>(x, xbf, n);
  }
  prep_lstm_w_kernel<<<2048, 256, 0, stream>>>(enc_k0, enc_r0, D_, WT0);
  prep_lstm_w_kernel<<<4096, 256, 0, stream>>>(enc_k1, enc_r1, H_, WT1);
  prep_lstm_w_kernel<<<4096, 256, 0, stream>>>(dec_k0, dec_r0, H_, WT2);
  prep_lstm_w_kernel<<<4096, 256, 0, stream>>>(dec_k1, dec_r1, H_, WT3);
  prep_dense_w_kernel<<<1024, 256, 0, stream>>>(enc_dw, H_, WTde);
  prep_dense_w_kernel<<<64, 256, 0, stream>>>(dec_dw, D_, WTdd);

  dim3 lgrid(8, 4), lblk(128);
  // encoder LSTMs
  lstm_layer_kernel<<<lgrid, lblk, 0, stream>>>(xbf, D_, WT0, enc_b0, seqA,
                                                hbuf, cnt + 0);
  lstm_layer_kernel<<<lgrid, lblk, 0, stream>>>(seqA, H_, WT1, enc_b1, seqB,
                                                hbuf, cnt + 4);
  // encoded = seqB @ enc_dw + enc_db  -> d_out (f32) + seqA (bf16 for decoder)
  dense_kernel<<<dim3(H_ / 16, (B_ * T_) / 16 / 8), 256, 0, stream>>>(
      seqB, WTde, enc_db, H_, outf, seqA);
  // decoder LSTMs
  lstm_layer_kernel<<<lgrid, lblk, 0, stream>>>(seqA, H_, WT2, dec_b0, seqB,
                                                hbuf, cnt + 8);
  lstm_layer_kernel<<<lgrid, lblk, 0, stream>>>(seqB, H_, WT3, dec_b1, seqA,
                                                hbuf, cnt + 12);
  // decoded = seqA @ dec_dw + dec_db -> d_out tail (f32)
  dense_kernel<<<dim3(D_ / 16, (B_ * T_) / 16 / 8), 256, 0, stream>>>(
      seqA, WTdd, dec_db, D_, outf + enc_elems, (bf16_t*)nullptr);
}
